// MLA_15410342658053
// MI455X (gfx1250) — matrix operand order repaired and checked by reading, not compiled, not run on hardware
//
#include <hip/hip_runtime.h>

// ---------------------------------------------------------------------------
// MLA-style block for gfx1250 (wave32):
//   qkv proj (bf16 WMMA GEMM, TDM-staged LDS) -> per-position 16x16 head
//   attention (VALU) -> scrambled reshape -> output proj (bf16 WMMA GEMM).
// ---------------------------------------------------------------------------

typedef __attribute__((ext_vector_type(16))) __bf16   v16bf;
typedef __attribute__((ext_vector_type(8)))  float    v8f;
typedef __attribute__((ext_vector_type(4)))  unsigned v4u;

struct Frag2 { v4u lo, hi; };

__device__ __forceinline__ float bf2f(unsigned short h) {
    unsigned u = ((unsigned)h) << 16;
    return __builtin_bit_cast(float, u);
}
__device__ __forceinline__ unsigned short f2bf(float f) {
    unsigned u = __builtin_bit_cast(unsigned, f);
    u += 0x7FFFu + ((u >> 16) & 1u);          // round-to-nearest-even
    return (unsigned short)(u >> 16);
}

// ---------------------------------------------------------------------------
// f32 -> bf16 convert
// ---------------------------------------------------------------------------
__global__ void __launch_bounds__(256)
conv_f32_bf16(const float* __restrict__ src, unsigned short* __restrict__ dst, int n) {
    int i = blockIdx.x * 256 + threadIdx.x;
    if (i < n) dst[i] = f2bf(src[i]);
}

// ---------------------------------------------------------------------------
// Tensor Data Mover: 2-D tile (tile_k x rows, bf16) global -> LDS, with
// LDS row padding 64B payload + 16B pad (matches LDS_STRIDE = 80).
// D# layout per CDNA5 ISA.
// This toolchain exposes the 6-arg builtin:
//   (uint32x4 g0, int32x8 g1, int32x4 g2, int32x4 g3, int32x8 g4, i32 cpol)
// groups 2/3/4 are zero for a 2-D tile.
// ---------------------------------------------------------------------------
#if __has_builtin(__builtin_amdgcn_tensor_load_to_lds)
#define USE_TDM 1
typedef __attribute__((ext_vector_type(4))) unsigned tdm_v4u;
typedef __attribute__((ext_vector_type(8))) int      tdm_v8i;
typedef __attribute__((ext_vector_type(4))) int      tdm_v4i;

__device__ __forceinline__ void tdm_load_tile_2d(const void* gaddr, unsigned lds_off,
                                                 int tile_k, int rows, int stride_elems)
{
    const unsigned long long ga = (unsigned long long)(uintptr_t)gaddr;
    // group0: count=1 | lds_addr | global_addr[56:0] | type=2
    tdm_v4u g0 = {
        1u,
        lds_off,
        (unsigned)(ga & 0xFFFFFFFFull),
        (unsigned)((ga >> 32) & 0x01FFFFFFull) | (2u << 30)
    };
    // group1: data_size=2B, pad_enable, pad_interval=16 dwords, pad_amount=4 dwords
    const unsigned w0 = (1u << 16) | (1u << 20) | (3u << 22) | (3u << 25);
    tdm_v8i g1 = {
        (int)w0,
        (int)(((unsigned)tile_k & 0xFFFFu) << 16),                       // tensor_dim0[15:0]
        (int)((((unsigned)tile_k >> 16) & 0xFFFFu) |
              (((unsigned)rows & 0xFFFFu) << 16)),                       // dim0 hi | tensor_dim1 lo
        (int)((((unsigned)rows >> 16) & 0xFFFFu) |
              (((unsigned)tile_k & 0xFFFFu) << 16)),                     // dim1 hi | tile_dim0
        (int)((unsigned)rows & 0xFFFFu),                                 // tile_dim1 | tile_dim2=0
        (int)(unsigned)stride_elems,                                     // tensor_dim0_stride[31:0]
        0, 0
    };
    tdm_v4i gz4 = {0, 0, 0, 0};
    tdm_v8i gz8 = {0, 0, 0, 0, 0, 0, 0, 0};
    __builtin_amdgcn_tensor_load_to_lds(g0, g1, gz4, gz4, gz8, 0);
}
#else
#define USE_TDM 0
#endif

// ---------------------------------------------------------------------------
// Y[M,N] = X[M,K] @ W[N,K]^T   (bf16 in, f32 accumulate)
// block = 128(M) x 256(N), BK=32, 256 threads = 8 waves as 2(M) x 4(N),
// each wave computes 64x64 = 4x4 tiles of 16x16 via WMMA bf16.
// Double-buffered LDS; tiles staged by TDM (or manual fallback).
// ---------------------------------------------------------------------------
#define LDS_STRIDE   80
#define TILE_A_BYTES (128 * LDS_STRIDE)          // 10240
#define TILE_B_BYTES (256 * LDS_STRIDE)          // 20480
#define BUF_BYTES    (TILE_A_BYTES + TILE_B_BYTES)

template <bool OUT_BF16>
__global__ void __launch_bounds__(256)
gemm_xwT(const unsigned short* __restrict__ A,   // [M,K] bf16
         const unsigned short* __restrict__ W,   // [N,K] bf16
         void* __restrict__ Cout,
         int N, int K)
{
    __shared__ __align__(16) unsigned char smem[2 * BUF_BYTES];   // 60 KB

    const int t    = threadIdx.x;
    const int lane = t & 31;
    const int wave = t >> 5;
    const int waveM = wave & 1;          // 0..1  -> 64-row slab
    const int waveN = wave >> 1;         // 0..3  -> 64-col slab
    const int half = lane >> 4;          // 0/1
    const int l16  = lane & 15;

    const int m0 = blockIdx.y * 128;
    const int n0 = blockIdx.x * 256;
    const int KT = K >> 5;               // K / 32

    v8f acc[4][4] = {};

#if USE_TDM
    const unsigned lds_base = (unsigned)(uintptr_t)(void*)smem;
    auto issue_tile = [&](int kt, int buf) {
        tdm_load_tile_2d(A + (size_t)m0 * K + kt * 32,
                         lds_base + buf * BUF_BYTES, 32, 128, K);
        tdm_load_tile_2d(W + (size_t)n0 * K + kt * 32,
                         lds_base + buf * BUF_BYTES + TILE_A_BYTES, 32, 256, K);
    };
    if (wave == 0) issue_tile(0, 0);
#else
    v4u regA[2], regB[4];
    auto load_regs = [&](int k0) {
        #pragma unroll
        for (int c = 0; c < 2; ++c) {
            const int ch = t + c * 256;
            regA[c] = *(const v4u*)(A + (size_t)(m0 + (ch >> 2)) * K + k0 + (ch & 3) * 8);
        }
        #pragma unroll
        for (int c = 0; c < 4; ++c) {
            const int ch = t + c * 256;
            regB[c] = *(const v4u*)(W + (size_t)(n0 + (ch >> 2)) * K + k0 + (ch & 3) * 8);
        }
    };
    auto store_regs = [&](int buf) {
        unsigned char* sA = smem + buf * BUF_BYTES;
        unsigned char* sB = sA + TILE_A_BYTES;
        #pragma unroll
        for (int c = 0; c < 2; ++c) {
            const int ch = t + c * 256;
            *(v4u*)(sA + (ch >> 2) * LDS_STRIDE + (ch & 3) * 16) = regA[c];
        }
        #pragma unroll
        for (int c = 0; c < 4; ++c) {
            const int ch = t + c * 256;
            *(v4u*)(sB + (ch >> 2) * LDS_STRIDE + (ch & 3) * 16) = regB[c];
        }
    };
    load_regs(0);
    store_regs(0);
    __syncthreads();
#endif

    for (int kt = 0; kt < KT; ++kt) {
        const int cur = kt & 1;
        const bool more = (kt + 1) < KT;

#if USE_TDM
        if (wave == 0) {
            if (more) {
                issue_tile(kt + 1, cur ^ 1);
                __builtin_amdgcn_s_wait_tensorcnt((short)2);   // tile kt landed
            } else {
                __builtin_amdgcn_s_wait_tensorcnt((short)0);
            }
            asm volatile("" ::: "memory");
        }
        __syncthreads();                   // publish tile kt to all waves
#else
        if (more) load_regs((kt + 1) << 5);
#endif

        const unsigned char* sA = smem + cur * BUF_BYTES;
        const unsigned char* sB = sA + TILE_A_BYTES;

        // A fragment: lane -> row m, half selects K-subblock; two b128
        v16bf afr[4];
        #pragma unroll
        for (int tm = 0; tm < 4; ++tm) {
            const int row = waveM * 64 + tm * 16 + l16;
            Frag2 p;
            p.lo = *(const v4u*)(sA + row * LDS_STRIDE + half * 16);
            p.hi = *(const v4u*)(sA + row * LDS_STRIDE + 32 + half * 16);
            afr[tm] = __builtin_bit_cast(v16bf, p);
        }
        // B fragment: lane -> column n, half selects K-subblock; two b128
        v16bf bfr[4];
        #pragma unroll
        for (int tn = 0; tn < 4; ++tn) {
            const int col = waveN * 64 + tn * 16 + l16;
            Frag2 p;
            p.lo = *(const v4u*)(sB + col * LDS_STRIDE + half * 16);
            p.hi = *(const v4u*)(sB + col * LDS_STRIDE + 32 + half * 16);
            bfr[tn] = __builtin_bit_cast(v16bf, p);
        }

        #pragma unroll
        for (int tm = 0; tm < 4; ++tm)
            #pragma unroll
            for (int tn = 0; tn < 4; ++tn)
                acc[tm][tn] = __builtin_amdgcn_wmma_f32_16x16x32_bf16(
                    false, afr[tm], false, bfr[tn],
                    (short)0, acc[tm][tn], false, false);

#if USE_TDM
        __syncthreads();                   // all done reading cur before overwrite
#else
        if (more) {
            __syncthreads();
            store_regs(cur ^ 1);
            __syncthreads();
        }
#endif
    }

    // epilogue: C/D layout: vgpr r -> m = r + 8*half, lane l16 -> n
    #pragma unroll
    for (int tm = 0; tm < 4; ++tm) {
        #pragma unroll
        for (int tn = 0; tn < 4; ++tn) {
            #pragma unroll
            for (int r = 0; r < 8; ++r) {
                const int row = m0 + waveM * 64 + tm * 16 + half * 8 + r;
                const int col = n0 + waveN * 64 + tn * 16 + l16;
                const size_t idx = (size_t)row * N + col;
                if (OUT_BF16)
                    ((unsigned short*)Cout)[idx] = f2bf(acc[tm][tn][r]);
                else
                    ((float*)Cout)[idx] = acc[tm][tn][r];
            }
        }
    }
}

// ---------------------------------------------------------------------------
// Per-position head attention. One workgroup per (b,s).
// qkv: [M, 6144] bf16 rows = [q(2048) | k(2048) | v(2048)], q,k,v as [16,128].
// Writes the reference's scrambled reshape directly:
//   Z[b, h*256 + s/16, (s%16)*128 + d] = out[b,s,h,d]   (bf16)
// ---------------------------------------------------------------------------
__global__ void __launch_bounds__(256)
attn_kernel(const unsigned short* __restrict__ qkv, unsigned short* __restrict__ Z)
{
    __shared__ float qs[16][128];
    __shared__ float ks[16][128];
    __shared__ float vs[16][128];
    __shared__ float sc[16][16];

    const int pos = blockIdx.x;          // b*4096 + s
    const int b = pos >> 12;
    const int s = pos & 4095;
    const unsigned short* base = qkv + (size_t)pos * 6144;

    const int t = threadIdx.x;
    const float scale = 0.08838834764831845f;   // 128^-0.5
    for (int i = t; i < 2048; i += 256) {
        qs[i >> 7][i & 127] = bf2f(base[i]) * scale;
        ks[i >> 7][i & 127] = bf2f(base[2048 + i]);
        vs[i >> 7][i & 127] = bf2f(base[4096 + i]);
    }
    __syncthreads();

    const int h  = t >> 4;
    const int tt = t & 15;

    float d = 0.f;
    #pragma unroll 8
    for (int kd = 0; kd < 128; ++kd) d += qs[h][kd] * ks[tt][kd];
    sc[h][tt] = d;
    __syncthreads();

    float mx = -3.4e38f;
    #pragma unroll
    for (int j = 0; j < 16; ++j) mx = fmaxf(mx, sc[h][j]);
    float sum = 0.f;
    #pragma unroll
    for (int j = 0; j < 16; ++j) sum += __expf(sc[h][j] - mx);
    const float a = __expf(d - mx) / sum;
    __syncthreads();
    sc[h][tt] = a;
    __syncthreads();

    const int dbase = tt * 8;
    float o[8] = {};
    #pragma unroll
    for (int j = 0; j < 16; ++j) {
        const float aw = sc[h][j];
        #pragma unroll
        for (int r = 0; r < 8; ++r) o[r] += aw * vs[j][dbase + r];
    }

    const int zr = h * 256 + (s >> 4);
    const int zc = (s & 15) * 128 + dbase;
    unsigned short* zp = Z + ((size_t)b * 4096 + zr) * 2048 + zc;
    #pragma unroll
    for (int r = 0; r < 8; ++r) zp[r] = f2bf(o[r]);
}

// ---------------------------------------------------------------------------
// Launcher
// ---------------------------------------------------------------------------
extern "C" void kernel_launch(void* const* d_in, const int* in_sizes, int n_in,
                              void* d_out, int out_size, void* d_ws, size_t ws_size,
                              hipStream_t stream)
{
    (void)in_sizes; (void)n_in; (void)out_size; (void)ws_size;

    constexpr int M = 4 * 4096;          // 16384
    constexpr int D = 2048;
    constexpr int NQKV = 3 * D;          // 6144
    constexpr int NX = M * D;            // 33554432
    constexpr int NW = D * D;            // 4194304

    const float* x  = (const float*)d_in[0];
    const float* wq = (const float*)d_in[1];
    const float* wk = (const float*)d_in[2];
    const float* wv = (const float*)d_in[3];
    const float* wo = (const float*)d_in[4];
    float* out = (float*)d_out;

    unsigned char* ws = (unsigned char*)d_ws;
    unsigned short* xb    = (unsigned short*)(ws);                         // 64MB
    unsigned short* wqkvb = (unsigned short*)(ws + 67108864ull);           // 24MB
    unsigned short* wob   = (unsigned short*)(ws + 92274688ull);           //  8MB
    unsigned short* qkvb  = (unsigned short*)(ws + 100663296ull);          // 192MB
    unsigned short* zb    = (unsigned short*)(ws + 301989888ull);          // 64MB

    conv_f32_bf16<<<NX / 256, 256, 0, stream>>>(x,  xb, NX);
    conv_f32_bf16<<<NW / 256, 256, 0, stream>>>(wq, wqkvb,           NW);
    conv_f32_bf16<<<NW / 256, 256, 0, stream>>>(wk, wqkvb + NW,      NW);
    conv_f32_bf16<<<NW / 256, 256, 0, stream>>>(wv, wqkvb + 2 * NW,  NW);
    conv_f32_bf16<<<NW / 256, 256, 0, stream>>>(wo, wob, NW);

    // qkv = x @ [wq;wk;wv]^T   -> [16384, 6144] bf16
    gemm_xwT<true><<<dim3(NQKV / 256, M / 128), 256, 0, stream>>>(
        xb, wqkvb, qkvb, NQKV, D);

    // per-position attention + scrambled reshape -> Z bf16 [16384, 2048]
    attn_kernel<<<M, 256, 0, stream>>>(qkvb, zb);

    // out = Z @ wo^T  -> f32 [16384, 2048]
    gemm_xwT<false><<<dim3(D / 256, M / 128), 256, 0, stream>>>(
        zb, wob, out, D, D);
}
